// CustomMultiLingualTransformer_6193342840930
// MI455X (gfx1250) — matrix operand order repaired and checked by reading, not compiled, not run on hardware
//
#include <hip/hip_runtime.h>

// ---------------------------------------------------------------------------
// CDNA5 / gfx1250 implementation of CustomMultiLingualTransformer forward.
// All GEMM-shaped work goes through v_wmma_f32_16x16x32_bf16 (wave32).
// Weights are pre-transposed + converted to bf16 [Npad][K]; GEMM B-tiles are
// staged with global_load_async_to_lds_b128 (ASYNCcnt) when available.
// ---------------------------------------------------------------------------

typedef __attribute__((ext_vector_type(16))) __bf16 v16bf;
typedef __attribute__((ext_vector_type(8)))  __bf16 v8bf;
typedef __attribute__((ext_vector_type(8)))  float  v8f;

#define WMMA_BF16(a, b, c) \
  __builtin_amdgcn_wmma_f32_16x16x32_bf16(false, (a), false, (b), (short)0, (c), false, false)

#define SH16(lo, hi) \
  __builtin_shufflevector((lo), (hi), 0, 1, 2, 3, 4, 5, 6, 7, 8, 9, 10, 11, 12, 13, 14, 15)

#if __has_builtin(__builtin_amdgcn_global_load_async_to_lds_b128) && \
    __has_builtin(__builtin_amdgcn_s_wait_asynccnt)
#define USE_ASYNC_LDS 1
typedef __attribute__((vector_size(16))) int v4i;
typedef __attribute__((address_space(1))) v4i* gv4i_p;   // global src
typedef __attribute__((address_space(3))) v4i* lv4i_p;   // LDS dst
#else
#define USE_ASYNC_LDS 0
#endif

constexpr int kB    = 4;
constexpr int kNCH  = 2;
constexpr int kSLEN = 8192;
constexpr int kD    = 512;
constexpr int kH    = 8;
constexpr int kDK   = 64;
constexpr int kFF   = 2048;
constexpr int kL    = 6;
constexpr int kV    = 32001;
constexpr int kVp   = 32064;         // kV padded to 64
constexpr int kT    = 1024;          // SLEN / 8
constexpr int kM    = kB * kT;       // 4096 rows

// ---------------------------------------------------------------------------
// Weight transpose+convert: Wt[n][k] (bf16, n zero-padded to Npad) from
// W[k][n] (f32). Coalesced writes; strided reads stay in L2.
// ---------------------------------------------------------------------------
__global__ void wt_bf16_kernel(const float* __restrict__ W, __bf16* __restrict__ Wt,
                               int K, int N, int total)
{
  const int idx = blockIdx.x * blockDim.x + threadIdx.x;
  if (idx >= total) return;
  const int k = idx % K;
  const int n = idx / K;
  Wt[idx] = (n < N) ? (__bf16)W[(size_t)k * N + n] : (__bf16)0.0f;
}

// ---------------------------------------------------------------------------
// Tiled GEMM: C[M,Nc] = A[M,K] @ Bt^T (+bias) (+resid) (+relu)
//   A : f32 row-major [M,K]        (M % 128 == 0, K % 32 == 0)
//   Bt: bf16 transposed [Npad,K]   (Npad = gridDim.x*64, zero-padded)
//   C : f32 row-major [M,Nc] with col<Nc store guard
// Block = 256 threads = 8 wave32s; block tile 128x64; K-step 32.
// Wave grid 4(M) x 2(N): each wave computes 32x32 = 4 WMMAs from
// 2 A-fragments and 2 B-fragments. No guarded staging paths.
// B staging: async DMA to LDS (no VGPR round-trip) when supported.
// ---------------------------------------------------------------------------
template<bool RELU, bool HAS_RESID>
__global__ __launch_bounds__(256) void gemm_bf16_kernel(
    const float* __restrict__ A, const __bf16* __restrict__ Bt,
    const float* __restrict__ bias, const float* __restrict__ resid,
    float* __restrict__ C, int M, int N, int K)
{
  __shared__ __bf16 As[128][40];    // [m][k]   row stride 80B (16B aligned)
  __shared__ __bf16 Bs[64][40];     // [n][k]

  const int tid  = threadIdx.x;
  const int lane = tid & 31;
  const int half = lane >> 4;       // 0 or 1 (wave half)
  const int l16  = lane & 15;
  const int wav  = tid >> 5;        // 0..7
  const int m0   = (wav >> 1) * 32; // 0,32,64,96
  const int n0   = (wav & 1) * 32;  // 0,32
  const int rowBase = blockIdx.y * 128;
  const int colBase = blockIdx.x * 64;
  const int abase   = half * 8;     // A fragment K base per ISA layout

  // staging maps
  const int arow  = tid >> 1;          // 0..127
  const int acol0 = (tid & 1) * 16;    // 0 or 16
  const int bn    = tid & 63;          // 0..63  (N row of Bt)
  const int kg    = (tid >> 6) * 8;    // 0,8,16,24 (K chunk)

  const float*  aBase = A  + (size_t)(rowBase + arow) * K;
  const __bf16* bBase = Bt + (size_t)(colBase + bn) * K;

  v8f acc00 = {}, acc01 = {}, acc10 = {}, acc11 = {};

  for (int k0 = 0; k0 < K; k0 += 32) {
    // ---- B tile: 16B DMA straight into LDS (ASYNCcnt) or sync fallback ----
#if USE_ASYNC_LDS
    __builtin_amdgcn_global_load_async_to_lds_b128(
        (gv4i_p)(bBase + k0 + kg), (lv4i_p)&Bs[bn][kg], 0, 0);
#else
    __builtin_prefetch(bBase + k0 + 32 + kg, 0, 1);
    *(v8bf*)&Bs[bn][kg] = *(const v8bf*)(bBase + k0 + kg);
#endif

    // ---- A tile: 16 consecutive floats per thread via 4x b128 loads ----
    const float4* ap = (const float4*)(aBase + k0 + acol0);
    __builtin_prefetch(aBase + k0 + 32 + acol0, 0, 1);
    const float4 a0 = ap[0], a1 = ap[1], a2 = ap[2], a3 = ap[3];
    v8bf p0, p1;
    p0[0] = (__bf16)a0.x; p0[1] = (__bf16)a0.y; p0[2] = (__bf16)a0.z; p0[3] = (__bf16)a0.w;
    p0[4] = (__bf16)a1.x; p0[5] = (__bf16)a1.y; p0[6] = (__bf16)a1.z; p0[7] = (__bf16)a1.w;
    p1[0] = (__bf16)a2.x; p1[1] = (__bf16)a2.y; p1[2] = (__bf16)a2.z; p1[3] = (__bf16)a2.w;
    p1[4] = (__bf16)a3.x; p1[5] = (__bf16)a3.y; p1[6] = (__bf16)a3.z; p1[7] = (__bf16)a3.w;
    *(v8bf*)&As[arow][acol0]     = p0;
    *(v8bf*)&As[arow][acol0 + 8] = p1;

#if USE_ASYNC_LDS
    __builtin_amdgcn_s_wait_asynccnt(0);   // drain this wave's DMA before barrier
#endif
    __syncthreads();

    // ---- fragments (all 16B contiguous groups per ISA layouts) ----
    const __bf16* a0p = &As[m0 + l16][abase];
    const __bf16* a1p = &As[m0 + 16 + l16][abase];
    const __bf16* b0p = &Bs[n0 + l16][half * 8];
    const __bf16* b1p = &Bs[n0 + 16 + l16][half * 8];
    const v16bf af0 = SH16(*(const v8bf*)a0p, *(const v8bf*)(a0p + 16));
    const v16bf af1 = SH16(*(const v8bf*)a1p, *(const v8bf*)(a1p + 16));
    const v16bf bf0 = SH16(*(const v8bf*)b0p, *(const v8bf*)(b0p + 16));
    const v16bf bf1 = SH16(*(const v8bf*)b1p, *(const v8bf*)(b1p + 16));

    acc00 = WMMA_BF16(af0, bf0, acc00);
    acc01 = WMMA_BF16(af0, bf1, acc01);
    acc10 = WMMA_BF16(af1, bf0, acc10);
    acc11 = WMMA_BF16(af1, bf1, acc11);
    __syncthreads();
  }

  // ---- epilogue: C layout -> VGPR r: rows r / 8+r per wave half ----
  const int c0 = colBase + n0 + l16;
  const int c1 = c0 + 16;
  const float bv0 = (c0 < N) ? bias[c0] : 0.0f;
  const float bv1 = (c1 < N) ? bias[c1] : 0.0f;
#pragma unroll
  for (int r = 0; r < 8; ++r) {
    const int rloc = half ? (8 + r) : r;
    const int r0 = rowBase + m0 + rloc;
    const int r1 = r0 + 16;
    if (c0 < N) {
      float v = acc00[r] + bv0;
      if (HAS_RESID) v += resid[(size_t)r0 * N + c0];
      if (RELU) v = fmaxf(v, 0.0f);
      C[(size_t)r0 * N + c0] = v;
      float w = acc10[r] + bv0;
      if (HAS_RESID) w += resid[(size_t)r1 * N + c0];
      if (RELU) w = fmaxf(w, 0.0f);
      C[(size_t)r1 * N + c0] = w;
    }
    if (c1 < N) {
      float v = acc01[r] + bv1;
      if (HAS_RESID) v += resid[(size_t)r0 * N + c1];
      if (RELU) v = fmaxf(v, 0.0f);
      C[(size_t)r0 * N + c1] = v;
      float w = acc11[r] + bv1;
      if (HAS_RESID) w += resid[(size_t)r1 * N + c1];
      if (RELU) w = fmaxf(w, 0.0f);
      C[(size_t)r1 * N + c1] = w;
    }
  }
}

// ---------------------------------------------------------------------------
// Repack f32 q/k/v into bf16: q,k row-major [B*T, 512]; v transposed to
// [b][h][dk][t] so attention's V fragments are contiguous in the key dim.
// ---------------------------------------------------------------------------
__global__ void pack_qkv_kernel(
    const float* __restrict__ q, const float* __restrict__ k,
    const float* __restrict__ v, __bf16* __restrict__ qbf,
    __bf16* __restrict__ kbf, __bf16* __restrict__ vtb)
{
  const int idx = blockIdx.x * blockDim.x + threadIdx.x;
  if (idx >= kM * kD) return;
  const int d = idx % kD;
  const int m = idx / kD;
  qbf[idx] = (__bf16)q[idx];
  kbf[idx] = (__bf16)k[idx];
  const int b = m / kT, t = m % kT;
  const int h = d / kDK, dk = d % kDK;
  vtb[(((size_t)b * kH + h) * kDK + dk) * kT + t] = (__bf16)v[idx];
}

// ---------------------------------------------------------------------------
// Attention: one wave per (b, h, 16-query tile). Scores via WMMA into a
// 32 KB bf16 LDS row buffer, full-row softmax, then P @ V via WMMA.
// All fragments are 16/32B contiguous loads from the bf16-packed buffers.
// ---------------------------------------------------------------------------
__global__ __launch_bounds__(32) void attn_kernel(
    const __bf16* __restrict__ qbf, const __bf16* __restrict__ kbf,
    const __bf16* __restrict__ vtb, const int* __restrict__ lens,
    float* __restrict__ ctx)
{
  __shared__ __bf16 Sb[16][kT];   // 32 KB

  const int qt   = blockIdx.x;    // 0..63
  const int h    = blockIdx.y;    // 0..7
  const int b    = blockIdx.z;    // 0..3
  const int lane = threadIdx.x & 31;
  const int half = lane >> 4;
  const int l16  = lane & 15;
  const int abase = half * 8;
  const int len  = lens[b];
  const float scale = 0.125f;     // 1/sqrt(64)

  // Q fragments for dk chunks [0,32) and [32,64)
  const int qrow = qt * 16 + l16;
  const __bf16* qp = qbf + ((size_t)(b * kT + qrow)) * kD + h * kDK;
  const v16bf aq0 = SH16(*(const v8bf*)(qp + abase),      *(const v8bf*)(qp + abase + 16));
  const v16bf aq1 = SH16(*(const v8bf*)(qp + 32 + abase), *(const v8bf*)(qp + 32 + abase + 16));

  // scores S = (Q K^T) * scale, masked, into LDS
  for (int kt = 0; kt < kT / 16; ++kt) {
    const int key = kt * 16 + l16;
    const __bf16* kp = kbf + ((size_t)(b * kT + key)) * kD + h * kDK + half * 8;
    const v16bf bk0 = SH16(*(const v8bf*)kp,        *(const v8bf*)(kp + 16));
    const v16bf bk1 = SH16(*(const v8bf*)(kp + 32), *(const v8bf*)(kp + 48));
    v8f c = {};
    c = WMMA_BF16(aq0, bk0, c);
    c = WMMA_BF16(aq1, bk1, c);
    const bool kvalid = (key < len);
#pragma unroll
    for (int r = 0; r < 8; ++r) {
      const int rloc = half ? (8 + r) : r;
      const float sv = kvalid ? (c[r] * scale) : -1.0e9f;
      Sb[rloc][kt * 16 + l16] = (__bf16)sv;
    }
  }
  __syncthreads();

  // row softmax (lanes L and L+16 redundantly own row L; split the writeback)
  {
    const int row = l16;
    float m = -3.0e38f;
    for (int j = 0; j < kT; ++j) m = fmaxf(m, (float)Sb[row][j]);
    float s = 0.0f;
    for (int j = 0; j < kT; ++j) s += __expf((float)Sb[row][j] - m);
    const float inv = 1.0f / s;
    const int kbeg = half * (kT / 2);
    for (int j = 0; j < kT / 2; ++j) {
      const int idx = kbeg + j;
      Sb[row][idx] = (__bf16)(__expf((float)Sb[row][idx] - m) * inv);
    }
  }
  __syncthreads();

  // O = P @ V : K-step = 32 keys, 4 N-chunks of 16 dk
  v8f acc[4] = {};
  for (int ks = 0; ks < kT / 32; ++ks) {
    const __bf16* sp = &Sb[l16][ks * 32 + abase];
    const v16bf pf = SH16(*(const v8bf*)sp, *(const v8bf*)(sp + 16));
#pragma unroll
    for (int nc = 0; nc < 4; ++nc) {
      const __bf16* vp = vtb + (((size_t)b * kH + h) * kDK + nc * 16 + l16) * kT
                             + ks * 32 + half * 8;
      const v16bf vf = SH16(*(const v8bf*)vp, *(const v8bf*)(vp + 16));
      acc[nc] = WMMA_BF16(pf, vf, acc[nc]);
    }
  }
#pragma unroll
  for (int nc = 0; nc < 4; ++nc) {
#pragma unroll
    for (int r = 0; r < 8; ++r) {
      const int rloc = half ? (8 + r) : r;
      const int row  = qt * 16 + rloc;
      ctx[((size_t)(b * kT + row)) * kD + h * kDK + nc * 16 + l16] = acc[nc][r];
    }
  }
}

// ---------------------------------------------------------------------------
// Conv1d (k=3, stride=2, pad=1) + BN-ish scale + ReLU.  out[b][co][t]
// ---------------------------------------------------------------------------
__global__ void conv_bn_relu_kernel(
    const float* __restrict__ in, int inBS, int Ci, int Ti,
    const float* __restrict__ w, const float* __restrict__ bias,
    const float* __restrict__ g, const float* __restrict__ beta,
    float* __restrict__ out, int Co, int To)
{
  const int idx = blockIdx.x * blockDim.x + threadIdx.x;
  const int total = kB * Co * To;
  if (idx >= total) return;
  const int t  = idx % To;
  const int co = (idx / To) % Co;
  const int b  = idx / (To * Co);
  float acc = bias[co];
  const int t0 = 2 * t - 1;
  for (int ci = 0; ci < Ci; ++ci) {
    const float* ip = in + (size_t)b * inBS + (size_t)ci * Ti;
    const float* wp = w + ((size_t)co * Ci + ci) * 3;
#pragma unroll
    for (int kk = 0; kk < 3; ++kk) {
      const int ti = t0 + kk;
      const float xv = (ti >= 0 && ti < Ti) ? ip[ti] : 0.0f;
      acc += wp[kk] * xv;
    }
  }
  const float y = g[co] * acc * 0.9999950000374996f + beta[co]; // 1/sqrt(1+1e-5)
  out[((size_t)b * Co + co) * To + t] = fmaxf(y, 0.0f);
}

// concat channel features (B,256,T) x2 -> x0[(b*T+t)*512 + ch*256 + c]
__global__ void pack_fusion_input_kernel(
    const float* __restrict__ f0, const float* __restrict__ f1,
    float* __restrict__ x0)
{
  const int idx = blockIdx.x * blockDim.x + threadIdx.x;
  if (idx >= kM * kD) return;
  const int d = idx % kD;
  const int m = idx / kD;
  const int b = m / kT, t = m % kT;
  const int ch = d / 256, c = d % 256;
  const float* f = ch ? f1 : f0;
  x0[idx] = f[((size_t)b * 256 + c) * kT + t];
}

// x += sinusoidal PE + lang embedding
__global__ void pe_lang_kernel(
    float* __restrict__ x, const int* __restrict__ langs,
    const float* __restrict__ lang_emb)
{
  const int idx = blockIdx.x * blockDim.x + threadIdx.x;
  if (idx >= kM * kD) return;
  const int d = idx % kD;
  const int m = idx / kD;
  const int b = m / kT;
  const float t = (float)(m % kT);
  const float freq = __expf((float)(d & ~1) * -0.017988946039015984f); // -ln(1e4)/512
  const float pe = (d & 1) ? __cosf(t * freq) : __sinf(t * freq);
  x[idx] += pe + lang_emb[(size_t)langs[b] * kD + d];
}

// LayerNorm over D=512; one block (128 threads) per row
__global__ __launch_bounds__(128) void ln_kernel(
    const float* __restrict__ in, const float* __restrict__ g,
    const float* __restrict__ bb, float* __restrict__ out)
{
  __shared__ float sm[4];
  const int row = blockIdx.x;
  const int tid = threadIdx.x;
  const float* p = in + (size_t)row * kD;
  float vals[4];
  float s = 0.0f;
#pragma unroll
  for (int j = 0; j < 4; ++j) { vals[j] = p[tid + j * 128]; s += vals[j]; }
#pragma unroll
  for (int off = 16; off > 0; off >>= 1) s += __shfl_xor(s, off, 32);
  if ((tid & 31) == 0) sm[tid >> 5] = s;
  __syncthreads();
  const float mean = (sm[0] + sm[1] + sm[2] + sm[3]) * (1.0f / kD);
  __syncthreads();
  float v2 = 0.0f;
#pragma unroll
  for (int j = 0; j < 4; ++j) { const float dd = vals[j] - mean; v2 += dd * dd; }
#pragma unroll
  for (int off = 16; off > 0; off >>= 1) v2 += __shfl_xor(v2, off, 32);
  if ((tid & 31) == 0) sm[tid >> 5] = v2;
  __syncthreads();
  const float var = (sm[0] + sm[1] + sm[2] + sm[3]) * (1.0f / kD);
  const float inv = rsqrtf(var + 1e-5f);
#pragma unroll
  for (int j = 0; j < 4; ++j) {
    const int c = tid + j * 128;
    out[(size_t)row * kD + c] = g[c] * (vals[j] - mean) * inv + bb[c];
  }
}

// in-place log_softmax over V=32001; one block (256 threads) per row
__global__ __launch_bounds__(256) void logsoftmax_kernel(float* __restrict__ p, int V)
{
  __shared__ float sm[8];
  const int row = blockIdx.x;
  float* x = p + (size_t)row * V;
  const int tid = threadIdx.x;

  float m = -3.0e38f;
  for (int i = tid; i < V; i += 256) m = fmaxf(m, x[i]);
#pragma unroll
  for (int off = 16; off > 0; off >>= 1) m = fmaxf(m, __shfl_xor(m, off, 32));
  if ((tid & 31) == 0) sm[tid >> 5] = m;
  __syncthreads();
  m = sm[0];
#pragma unroll
  for (int j = 1; j < 8; ++j) m = fmaxf(m, sm[j]);
  __syncthreads();

  float s = 0.0f;
  for (int i = tid; i < V; i += 256) s += __expf(x[i] - m);
#pragma unroll
  for (int off = 16; off > 0; off >>= 1) s += __shfl_xor(s, off, 32);
  if ((tid & 31) == 0) sm[tid >> 5] = s;
  __syncthreads();
  s = sm[0] + sm[1] + sm[2] + sm[3] + sm[4] + sm[5] + sm[6] + sm[7];
  const float lse = m + __logf(s);
  for (int i = tid; i < V; i += 256) x[i] -= lse;
}

// ---------------------------------------------------------------------------
// Host-side orchestration.
// Input flattening assumption: jax pytree order (dict keys sorted):
//   [0]=audio, [1]=audio_lengths, [2]=language_ids, then params:
//   convs (2ch x 3layers x {b,beta,g,w}) = 24 leaves,
//   enc (6 x {ff1.b,ff1.w,ff2.b,ff2.w,ln1_b,ln1_g,ln2_b,ln2_g,
//             wk.b,wk.w,wo.b,wo.w,wq.b,wq.w,wv.b,wv.w}) = 96 leaves,
//   fusion.b, fusion.w, lang_emb, out.b, out.w.
// Workspace: ~118 MB f32 scratch + ~84 MB bf16 (qkv packs + weights).
// ---------------------------------------------------------------------------
extern "C" void kernel_launch(void* const* d_in, const int* in_sizes, int n_in,
                              void* d_out, int out_size, void* d_ws, size_t ws_size,
                              hipStream_t stream)
{
  (void)in_sizes; (void)n_in; (void)out_size; (void)ws_size;

  const float* audio = (const float*)d_in[0];
  const int*   lens  = (const int*)d_in[1];
  const int*   langs = (const int*)d_in[2];

  int pi = 3;
  const float* convP[2][3][4]; // [ch][layer][b,beta,g,w]
  for (int ch = 0; ch < 2; ++ch)
    for (int l = 0; l < 3; ++l)
      for (int f = 0; f < 4; ++f) convP[ch][l][f] = (const float*)d_in[pi++];

  const float* encP[kL][16];
  for (int l = 0; l < kL; ++l)
    for (int f = 0; f < 16; ++f) encP[l][f] = (const float*)d_in[pi++];
  // enc field order: 0 ff1.b 1 ff1.w 2 ff2.b 3 ff2.w 4 ln1_b 5 ln1_g 6 ln2_b
  //                  7 ln2_g 8 wk.b 9 wk.w 10 wo.b 11 wo.w 12 wq.b 13 wq.w
  //                  14 wv.b 15 wv.w

  const float* fusion_b = (const float*)d_in[pi++];
  const float* fusion_w = (const float*)d_in[pi++];
  const float* lang_emb = (const float*)d_in[pi++];
  const float* out_b    = (const float*)d_in[pi++];
  const float* out_w    = (const float*)d_in[pi++];

  // ---- workspace partition ----
  float* ws  = (float*)d_ws;
  float* c0  = ws;                                  // B*64*4096  = 1,048,576
  float* c1  = c0  + (size_t)1048576;               // B*128*2048 = 1,048,576
  float* f0  = c1  + (size_t)1048576;               // B*256*1024 = 1,048,576
  float* f1  = f0  + (size_t)1048576;
  float* x0  = f1  + (size_t)1048576;               // 4096*512
  float* x   = x0  + (size_t)kM * kD;
  float* q   = x   + (size_t)kM * kD;
  float* kbuf= q   + (size_t)kM * kD;
  float* vbuf= kbuf+ (size_t)kM * kD;
  float* ctx = vbuf+ (size_t)kM * kD;
  float* t1  = ctx + (size_t)kM * kD;
  float* aa  = t1  + (size_t)kM * kD;
  float* ff  = aa  + (size_t)kM * kD;               // 4096*2048

  __bf16* bfp = (__bf16*)(ff + (size_t)kM * kFF);
  __bf16* qbf = bfp;                 bfp += (size_t)kM * kD;
  __bf16* kbf = bfp;                 bfp += (size_t)kM * kD;
  __bf16* vtb = bfp;                 bfp += (size_t)kM * kD;
  __bf16* fusion_t = bfp;            bfp += (size_t)kD * kD;
  __bf16* encT[kL][6];               // wq,wk,wv,wo,ff1,ff2 (transposed bf16)
  for (int l = 0; l < kL; ++l) {
    for (int j = 0; j < 4; ++j) { encT[l][j] = bfp; bfp += (size_t)kD * kD; }
    encT[l][4] = bfp;               bfp += (size_t)kFF * kD;   // ff1: [2048][512]
    encT[l][5] = bfp;               bfp += (size_t)kD * kFF;   // ff2: [512][2048]
  }
  __bf16* out_t = bfp;              bfp += (size_t)kVp * kD;   // [32064][512]

  auto cvt_w = [&](const float* W, __bf16* Wt, int K, int N) {
    const int Npad  = (N + 63) & ~63;
    const int total = Npad * K;
    wt_bf16_kernel<<<(total + 255) / 256, 256, 0, stream>>>(W, Wt, K, N, total);
  };

  auto gemm = [&](const float* A, const __bf16* Bt, const float* bias,
                  const float* resid, float* C, int M, int N, int K, bool relu) {
    dim3 grid((N + 63) / 64, (M + 127) / 128);
    if (resid)
      gemm_bf16_kernel<false, true><<<grid, 256, 0, stream>>>(A, Bt, bias, resid, C, M, N, K);
    else if (relu)
      gemm_bf16_kernel<true, false><<<grid, 256, 0, stream>>>(A, Bt, bias, nullptr, C, M, N, K);
    else
      gemm_bf16_kernel<false, false><<<grid, 256, 0, stream>>>(A, Bt, bias, nullptr, C, M, N, K);
  };

  // ---- pre-transpose/convert all weights to bf16 [Npad][K] ----
  cvt_w(fusion_w, fusion_t, kD, kD);
  for (int l = 0; l < kL; ++l) {
    cvt_w(encP[l][13], encT[l][0], kD, kD);    // wq
    cvt_w(encP[l][9],  encT[l][1], kD, kD);    // wk
    cvt_w(encP[l][15], encT[l][2], kD, kD);    // wv
    cvt_w(encP[l][11], encT[l][3], kD, kD);    // wo
    cvt_w(encP[l][1],  encT[l][4], kD, kFF);   // ff1
    cvt_w(encP[l][3],  encT[l][5], kFF, kD);   // ff2
  }
  cvt_w(out_w, out_t, kD, kV);

  // ---- conv frontend (per channel) ----
  for (int ch = 0; ch < 2; ++ch) {
    const float* in0 = audio + (size_t)ch * kSLEN;  // batch stride NCH*SLEN
    int total = kB * 64 * 4096;
    conv_bn_relu_kernel<<<(total + 255) / 256, 256, 0, stream>>>(
        in0, kNCH * kSLEN, 1, kSLEN,
        convP[ch][0][3], convP[ch][0][0], convP[ch][0][2], convP[ch][0][1],
        c0, 64, 4096);
    total = kB * 128 * 2048;
    conv_bn_relu_kernel<<<(total + 255) / 256, 256, 0, stream>>>(
        c0, 64 * 4096, 64, 4096,
        convP[ch][1][3], convP[ch][1][0], convP[ch][1][2], convP[ch][1][1],
        c1, 128, 2048);
    total = kB * 256 * 1024;
    conv_bn_relu_kernel<<<(total + 255) / 256, 256, 0, stream>>>(
        c1, 128 * 2048, 128, 2048,
        convP[ch][2][3], convP[ch][2][0], convP[ch][2][2], convP[ch][2][1],
        ch ? f1 : f0, 256, 1024);
  }

  // ---- fusion linear + positional/lang embedding ----
  {
    const int total = kM * kD;
    pack_fusion_input_kernel<<<(total + 255) / 256, 256, 0, stream>>>(f0, f1, x0);
  }
  gemm(x0, fusion_t, fusion_b, nullptr, x, kM, kD, kD, false);
  {
    const int total = kM * kD;
    pe_lang_kernel<<<(total + 255) / 256, 256, 0, stream>>>(x, langs, lang_emb);
  }

  // ---- encoder layers ----
  for (int l = 0; l < kL; ++l) {
    gemm(x, encT[l][0], encP[l][12], nullptr, q,    kM, kD, kD, false); // wq
    gemm(x, encT[l][1], encP[l][8],  nullptr, kbuf, kM, kD, kD, false); // wk
    gemm(x, encT[l][2], encP[l][14], nullptr, vbuf, kM, kD, kD, false); // wv

    {
      const int total = kM * kD;
      pack_qkv_kernel<<<(total + 255) / 256, 256, 0, stream>>>(
          q, kbuf, vbuf, qbf, kbf, vtb);
    }
    attn_kernel<<<dim3(kT / 16, kH, kB), 32, 0, stream>>>(qbf, kbf, vtb, lens, ctx);

    gemm(ctx, encT[l][3], encP[l][10], x, t1, kM, kD, kD, false);       // wo + resid
    ln_kernel<<<kM, 128, 0, stream>>>(t1, encP[l][5], encP[l][4], aa);  // ln1

    gemm(aa, encT[l][4], encP[l][0], nullptr, ff, kM, kFF, kD, true);   // ff1 + relu
    gemm(ff, encT[l][5], encP[l][2], aa, t1, kM, kD, kFF, false);       // ff2 + resid
    ln_kernel<<<kM, 128, 0, stream>>>(t1, encP[l][7], encP[l][6], x);   // ln2 -> x
  }

  // ---- output projection + in-place log_softmax on d_out ----
  gemm(x, out_t, out_b, nullptr, (float*)d_out, kM, kV, kD, false);
  logsoftmax_kernel<<<kM, 256, 0, stream>>>((float*)d_out, kV);
}
